// model_41970420417850
// MI455X (gfx1250) — matrix operand order repaired and checked by reading, not compiled, not run on hardware
//
#include <hip/hip_runtime.h>
#include <cstddef>
#include <cstdint>

// Problem constants (from the reference)
#define BB      8
#define LL      512
#define DD      768
#define NLAYERS 12
#define EE      8192
#define HH      12
#define DHD     64
#define DFF     3072
#define MTOK    (BB * LL)   // 4096 tokens

typedef __attribute__((ext_vector_type(16))) __bf16 v16bf;
typedef __attribute__((ext_vector_type(4)))  __bf16 v4bf;
typedef __attribute__((ext_vector_type(8)))  float  v8f;

struct U4 { unsigned a, b, c, d; };    // 16-byte chunk for LDS->VGPR fragment loads

__device__ __forceinline__ float gelu_f(float x) {
  return 0.5f * x * (1.0f + erff(x * 0.70710678118654752440f));
}

#define FLAG_BIAS 1
#define FLAG_GELU 2
#define FLAG_RES  4

#define TK   32
#define KPAD 40   // 80B row pitch: keeps 16B alignment, dodges bank conflicts

// ---------------------------------------------------------------------------
// Batched GEMM, guard-free (all shapes in this model are tile-aligned):
//   C = alpha*(A @ B) [+bias][+residual][gelu],  bf16 WMMA, f32 accumulate
// A: [M,K] row-major fp32.  B: [K,N] row-major, or [N,K] if b_is_nk.
// Block: 256 thr = 8 waves (4 M x 2 N).  Block tile 128 x BN.
// Wave tile 32 x (BN/2) -> 2 x (BN/32) WMMA fragments per K-step.
// fp32 -> bf16 conversion uses the native hardware converts (v_cvt_*_bf16).
// ---------------------------------------------------------------------------
template <int BN>
__global__ __launch_bounds__(256)
void gemm_bf16_wmma(const float* __restrict__ A, const float* __restrict__ Bm,
                    const float* __restrict__ bias, const float* __restrict__ Res,
                    float* __restrict__ C,
                    int K, int lda, int ldb, int ldc,
                    long long sA, long long sB, long long sC, long long sRes,
                    int b_is_nk, float alpha, int flags) {
  constexpr int BM = 128;
  constexpr int NF = BN / 32;          // N fragments per wave
  __shared__ __align__(16) __bf16 As[BM][KPAD];
  __shared__ __align__(16) __bf16 Bs[BN][KPAD];

  const int z = blockIdx.z;
  const float* Ab = A + (long long)z * sA;
  const float* Bb = Bm + (long long)z * sB;
  float*       Cb = C + (long long)z * sC;
  const float* Rb = Res ? (Res + (long long)z * sRes) : nullptr;

  const int tid  = threadIdx.x;
  const int lane = tid & 31;
  const int wid  = tid >> 5;
  const int wm   = wid >> 1;           // 0..3 -> 32-row slice
  const int wn   = wid & 1;            // 0..1 -> (BN/2)-col slice
  const int m0   = blockIdx.y * BM;
  const int n0   = blockIdx.x * BN;

  v8f zero = {};
  v8f acc[2][NF];
#pragma unroll
  for (int i = 0; i < 2; ++i)
#pragma unroll
    for (int j = 0; j < NF; ++j) acc[i][j] = zero;

  const int l15 = lane & 15;
  const int ah  = (lane >> 4) * 8;    // A frag: lanes<16 K {0..7,16..23}; lanes>=16 {8..15,24..31}
  const int bh  = (lane >> 4) * 8;    // B frag: lanes<16 K {0..7,16..23}; lanes>=16 {8..15,24..31}

  for (int k0 = 0; k0 < K; k0 += TK) {
    // ---- Stage A tile (BM x TK): b128 load -> native cvt -> packed b64 store ----
#pragma unroll
    for (int i = 0; i < (BM * TK) / (256 * 4); ++i) {
      int idx = tid + i * 256;
      int r   = idx >> 3;            // 8 float4-chunks per 32-wide row
      int c4  = (idx & 7) << 2;
      const float* src = &Ab[(long long)(m0 + r) * lda + k0 + c4];
      float4 v = *(const float4*)src;
      __builtin_prefetch(src + TK, 0, 1);        // speculative, next K-tile
      v4bf p;
      p.x = (__bf16)v.x; p.y = (__bf16)v.y; p.z = (__bf16)v.z; p.w = (__bf16)v.w;
      *(v4bf*)&As[r][c4] = p;
    }
    // ---- Stage B tile (BN x TK), stored [n][k] ----
    if (b_is_nk) {                   // B already [N][K]: straight vector copy
#pragma unroll
      for (int i = 0; i < (BN * TK) / (256 * 4); ++i) {
        int idx = tid + i * 256;
        int n   = idx >> 3;
        int c4  = (idx & 7) << 2;
        const float* src = &Bb[(long long)(n0 + n) * ldb + k0 + c4];
        float4 v = *(const float4*)src;
        __builtin_prefetch(src + TK, 0, 1);
        v4bf p;
        p.x = (__bf16)v.x; p.y = (__bf16)v.y; p.z = (__bf16)v.z; p.w = (__bf16)v.w;
        *(v4bf*)&Bs[n][c4] = p;
      }
    } else {                         // B is [K][N]: 4 coalesced row loads -> 1 b64 store
#pragma unroll
      for (int i = 0; i < (BN * TK) / (256 * 4); ++i) {
        int idx = tid + i * 256;
        int n   = idx % BN;
        int k4  = (idx / BN) << 2;
        const float* src = &Bb[(long long)(k0 + k4) * ldb + n0 + n];
        float a0 = src[0];
        float a1 = src[ldb];
        float a2 = src[2 * (long long)ldb];
        float a3 = src[3 * (long long)ldb];
        v4bf p;
        p.x = (__bf16)a0; p.y = (__bf16)a1; p.z = (__bf16)a2; p.w = (__bf16)a3;
        *(v4bf*)&Bs[n][k4] = p;
      }
    }
    __syncthreads();

    // ---- Fragments from LDS, 8 (or 4) WMMAs per K-step ----
    v16bf af[2];
#pragma unroll
    for (int i = 0; i < 2; ++i) {
      int mr = wm * 32 + i * 16 + l15;
      ((U4*)&af[i])[0] = *(const U4*)&As[mr][ah];
      ((U4*)&af[i])[1] = *(const U4*)&As[mr][ah + 16];
    }
    v16bf bfr[NF];
#pragma unroll
    for (int j = 0; j < NF; ++j) {
      int nc = wn * (BN / 2) + j * 16 + l15;
      ((U4*)&bfr[j])[0] = *(const U4*)&Bs[nc][bh];
      ((U4*)&bfr[j])[1] = *(const U4*)&Bs[nc][bh + 16];
    }
#pragma unroll
    for (int i = 0; i < 2; ++i)
#pragma unroll
      for (int j = 0; j < NF; ++j)
        acc[i][j] = __builtin_amdgcn_wmma_f32_16x16x32_bf16(
            false, af[i], false, bfr[j], (short)0, acc[i][j], false, false);
    __syncthreads();
  }

  // ---- Epilogue: VGPR r -> M = r + 8*(lane>=16); N = lane&15 ----
#pragma unroll
  for (int i = 0; i < 2; ++i) {
#pragma unroll
    for (int j = 0; j < NF; ++j) {
      int row0 = m0 + wm * 32 + i * 16 + ((lane >> 4) << 3);
      int col  = n0 + wn * (BN / 2) + j * 16 + l15;
      float bv = (flags & FLAG_BIAS) ? bias[col] : 0.0f;
#pragma unroll
      for (int r = 0; r < 8; ++r) {
        int row = row0 + r;
        float v = acc[i][j][r] * alpha + bv;
        if (flags & FLAG_RES)  v += Rb[(long long)row * ldc + col];
        if (flags & FLAG_GELU) v = gelu_f(v);
        Cb[(long long)row * ldc + col] = v;
      }
    }
  }
}

// ---------------------------------------------------------------------------
// Row softmax over 512 columns (one 128-thread block per row), in place.
// ---------------------------------------------------------------------------
__global__ __launch_bounds__(128)
void softmax512(float* __restrict__ S) {
  __shared__ float red[128];
  int t = threadIdx.x;
  float* row = S + (long long)blockIdx.x * LL;
  float a0 = row[t], a1 = row[t + 128], a2 = row[t + 256], a3 = row[t + 384];
  float m = fmaxf(fmaxf(a0, a1), fmaxf(a2, a3));
  red[t] = m; __syncthreads();
  for (int s = 64; s > 0; s >>= 1) { if (t < s) red[t] = fmaxf(red[t], red[t + s]); __syncthreads(); }
  m = red[0]; __syncthreads();
  a0 = __expf(a0 - m); a1 = __expf(a1 - m); a2 = __expf(a2 - m); a3 = __expf(a3 - m);
  red[t] = a0 + a1 + a2 + a3; __syncthreads();
  for (int s = 64; s > 0; s >>= 1) { if (t < s) red[t] += red[t + s]; __syncthreads(); }
  float inv = 1.0f / red[0];
  row[t] = a0 * inv; row[t + 128] = a1 * inv; row[t + 256] = a2 * inv; row[t + 384] = a3 * inv;
}

// ---------------------------------------------------------------------------
// LayerNorm over 768 columns (one 256-thread block per row), eps = 1e-12.
// ---------------------------------------------------------------------------
__global__ __launch_bounds__(256)
void layernorm768(const float* __restrict__ X, const float* __restrict__ sc,
                  const float* __restrict__ bs, float* __restrict__ Out) {
  __shared__ float r1[256], r2[256];
  int t = threadIdx.x;
  const float* row = X + (long long)blockIdx.x * DD;
  float x0 = row[t], x1 = row[t + 256], x2 = row[t + 512];
  r1[t] = x0 + x1 + x2;
  r2[t] = x0 * x0 + x1 * x1 + x2 * x2;
  __syncthreads();
  for (int s = 128; s > 0; s >>= 1) {
    if (t < s) { r1[t] += r1[t + s]; r2[t] += r2[t + s]; }
    __syncthreads();
  }
  float mean = r1[0] * (1.0f / 768.0f);
  float var  = r2[0] * (1.0f / 768.0f) - mean * mean;
  float inv  = rsqrtf(var + 1e-12f);
  float* orow = Out + (long long)blockIdx.x * DD;
  orow[t]       = (x0 - mean) * inv * sc[t]       + bs[t];
  orow[t + 256] = (x1 - mean) * inv * sc[t + 256] + bs[t + 256];
  orow[t + 512] = (x2 - mean) * inv * sc[t + 512] + bs[t + 512];
}

// ---------------------------------------------------------------------------
// QKV pack: [B,L,3D] -> q/k/v [B,H,L,DH]; ctx unpack: [B,H,L,DH] -> [B,L,D]
// ---------------------------------------------------------------------------
__global__ void pack_qkv_k(const float* __restrict__ qkv, float* __restrict__ q,
                           float* __restrict__ k, float* __restrict__ v) {
  int idx = blockIdx.x * blockDim.x + threadIdx.x;
  if (idx >= MTOK * DD) return;
  int d  = idx & (DHD - 1);
  int t1 = idx >> 6;
  int hh = t1 % HH;
  int t2 = t1 / HH;
  int l  = t2 & (LL - 1);
  int b  = t2 >> 9;
  long long srow = (long long)(b * LL + l) * (3 * DD);
  long long dsti = ((long long)(b * HH + hh) * LL + l) * DHD + d;
  q[dsti] = qkv[srow +          hh * DHD + d];
  k[dsti] = qkv[srow + DD +     hh * DHD + d];
  v[dsti] = qkv[srow + 2 * DD + hh * DHD + d];
}

__global__ void unpack_ctx_k(const float* __restrict__ ctxp, float* __restrict__ ctx) {
  int idx = blockIdx.x * blockDim.x + threadIdx.x;
  if (idx >= MTOK * DD) return;
  int d  = idx & (DHD - 1);
  int t1 = idx >> 6;
  int hh = t1 % HH;
  int t2 = t1 / HH;
  int l  = t2 & (LL - 1);
  int b  = t2 >> 9;
  ctx[(long long)(b * LL + l) * DD + hh * DHD + d] =
      ctxp[((long long)(b * HH + hh) * LL + l) * DHD + d];
}

// ---------------------------------------------------------------------------
// GCN helpers
// ---------------------------------------------------------------------------
__global__ void fill1_k(float* p, int n) {
  int i = blockIdx.x * blockDim.x + threadIdx.x;
  if (i < n) p[i] = 1.0f;   // self-loop contributes 1 to every degree
}

__global__ void deg_scatter_k(const int* __restrict__ ei, float* __restrict__ deg) {
  int i = blockIdx.x * blockDim.x + threadIdx.x;
  if (i >= BB * EE) return;
  int b = i / EE, e = i % EE;
  int d = ei[(long long)b * 2 * EE + EE + e];   // dst row
  atomicAdd(&deg[b * LL + d], 1.0f);
}

__global__ void deg_fin_k(float* p, int n) {
  int i = blockIdx.x * blockDim.x + threadIdx.x;
  if (i < n) p[i] = rsqrtf(fmaxf(p[i], 1.0f));  // deg -> deg^{-1/2}
}

__global__ void gcn_self_k(const float* __restrict__ xw, const float* __restrict__ dinv,
                           float* __restrict__ out) {
  int i = blockIdx.x * blockDim.x + threadIdx.x;
  if (i >= MTOK * DD) return;
  int row = i / DD;
  float di = dinv[row];
  out[i] = xw[i] * di * di;   // self-loop term initializes the accumulator
}

__global__ void gcn_edge_k(const float* __restrict__ xw, const int* __restrict__ ei,
                           const float* __restrict__ dinv, float* __restrict__ out) {
  long long i = (long long)blockIdx.x * blockDim.x + threadIdx.x;
  const long long total = (long long)BB * EE * (DD / 4);
  if (i >= total) return;
  int c = (int)(i % (DD / 4));
  long long t = i / (DD / 4);
  int e = (int)(t % EE);
  int b = (int)(t / EE);
  const int* eb = ei + (long long)b * 2 * EE;
  int s = eb[e], d = eb[EE + e];
  float co = dinv[b * LL + s] * dinv[b * LL + d];
  const float* xr = xw + (long long)(b * LL + s) * DD + c * 4;
  float* orw      = out + (long long)(b * LL + d) * DD + c * 4;
  atomicAdd(&orw[0], xr[0] * co);
  atomicAdd(&orw[1], xr[1] * co);
  atomicAdd(&orw[2], xr[2] * co);
  atomicAdd(&orw[3], xr[3] * co);
}

__global__ void bias_gelu_k(const float* __restrict__ in, const float* __restrict__ bsc,
                            float* __restrict__ out, int n, int ncols) {
  int i = blockIdx.x * blockDim.x + threadIdx.x;
  if (i >= n) return;
  out[i] = gelu_f(in[i] + bsc[i % ncols]);
}

__global__ void add_k(const float* __restrict__ a, const float* __restrict__ b,
                      float* __restrict__ c, int n) {
  int i = blockIdx.x * blockDim.x + threadIdx.x;
  if (i < n) c[i] = a[i] + b[i];
}

// ---------------------------------------------------------------------------
// Host orchestration
// ---------------------------------------------------------------------------
extern "C" void kernel_launch(void* const* d_in, const int* in_sizes, int n_in,
                              void* d_out, int out_size, void* d_ws, size_t ws_size,
                              hipStream_t stream) {
  (void)in_sizes; (void)n_in; (void)out_size;

  const float* hs0 = (const float*)d_in[0];
  const int*   ei  = (const int*)d_in[1];
  const float* P[28];
  for (int i = 2; i < 28; ++i) P[i] = (const float*)d_in[i];

  // Per-layer element strides (leading N=12 dim in each param)
  static const long long psz[28] = {
      0, 0,
      (long long)DD * 3 * DD, 3 * DD,            // Wqkv, bqkv
      (long long)DD * DD, DD,                    // Wo, bo
      DD, DD, DD, DD,                            // ln1_s, ln1_b, n1_s, n1_b
      (long long)DD * DD, DD,                    // m1_W, m1_b
      (long long)DD * DD, DD,                    // c1_W, c1_b
      (long long)DD * DD, DD,                    // c2_W, c2_b
      DD, DD,                                    // n2_s, n2_b
      (long long)DD * DD, DD,                    // m2_W, m2_b
      (long long)DD * DD, DD,                    // m3_W, m3_b
      (long long)DD * DFF, DFF,                  // int_W, int_b
      (long long)DFF * DD, DD,                   // out_W, out_b
      DD, DD                                     // ln2_s, ln2_b
  };
#define PP(i) (P[i] + (long long)layer * psz[i])

  // Workspace layout
  float* w = (float*)d_ws;
  size_t off = 0;
  auto alloc = [&](size_t nfl) { float* p = w + off; off += nfl; return p; };
  float* h    = alloc((size_t)MTOK * DD);
  float* qkv  = alloc((size_t)MTOK * 3 * DD);
  float* qp   = alloc((size_t)MTOK * DD);
  float* kp   = alloc((size_t)MTOK * DD);
  float* vp   = alloc((size_t)MTOK * DD);
  float* S    = alloc((size_t)BB * HH * LL * LL);
  float* ctxp = alloc((size_t)MTOK * DD);
  float* ctx  = alloc((size_t)MTOK * DD);
  float* tmp  = alloc((size_t)MTOK * DD);
  float* h1   = alloc((size_t)MTOK * DD);
  float* res  = alloc((size_t)MTOK * DD);
  float* x1   = alloc((size_t)MTOK * DD);
  float* x2   = alloc((size_t)MTOK * DD);
  float* xw   = alloc((size_t)MTOK * DD);
  float* tln  = alloc((size_t)MTOK * DD);
  float* g    = alloc((size_t)MTOK * DD);
  float* ffm  = alloc((size_t)MTOK * DFF);
  float* dinv = alloc((size_t)BB * LL);
  if (ws_size < off * sizeof(float)) return;   // not enough scratch; bail safely

  // All shapes are tile-aligned: M % 128 == 0, K % 32 == 0, N % 128 == 0 or N == 64.
  auto gemm = [&](const float* Am, const float* Bmm, const float* bias, const float* Rm,
                  float* Cm, int M, int N, int K, int lda, int ldb, int ldc,
                  long long sA, long long sB, long long sC, long long sR,
                  int bnk, float alpha, int flags, int batches) {
    if (N == 64) {
      dim3 gr(1u, (unsigned)(M / 128), (unsigned)batches);
      gemm_bf16_wmma<64><<<gr, 256, 0, stream>>>(Am, Bmm, bias, Rm, Cm, K,
                                                 lda, ldb, ldc, sA, sB, sC, sR,
                                                 bnk, alpha, flags);
    } else {
      dim3 gr((unsigned)(N / 128), (unsigned)(M / 128), (unsigned)batches);
      gemm_bf16_wmma<128><<<gr, 256, 0, stream>>>(Am, Bmm, bias, Rm, Cm, K,
                                                  lda, ldb, ldc, sA, sB, sC, sR,
                                                  bnk, alpha, flags);
    }
  };

  const int nBL = MTOK * DD;
  const int blkBL = (nBL + 255) / 256;

  // Symmetric-normalized degree (same for every layer): dinv = rsqrt(deg+1)
  {
    int n = BB * LL;
    fill1_k<<<(n + 255) / 256, 256, 0, stream>>>(dinv, n);
    int ne = BB * EE;
    deg_scatter_k<<<(ne + 255) / 256, 256, 0, stream>>>(ei, dinv);
    deg_fin_k<<<(n + 255) / 256, 256, 0, stream>>>(dinv, n);
  }
  hipMemcpyAsync(h, hs0, (size_t)nBL * sizeof(float), hipMemcpyDeviceToDevice, stream);

  for (int layer = 0; layer < NLAYERS; ++layer) {
    // ---- attention ----
    gemm(h, PP(2), PP(3), nullptr, qkv, MTOK, 3 * DD, DD, DD, 3 * DD, 3 * DD,
         0, 0, 0, 0, 0, 1.0f, FLAG_BIAS, 1);
    pack_qkv_k<<<blkBL, 256, 0, stream>>>(qkv, qp, kp, vp);
    // S = (Q @ K^T) / sqrt(DH): K stored [kk][d] == [N][K] for this GEMM
    gemm(qp, kp, nullptr, nullptr, S, LL, LL, DHD, DHD, DHD, LL,
         (long long)LL * DHD, (long long)LL * DHD, (long long)LL * LL, 0,
         1, 0.125f, 0, BB * HH);
    softmax512<<<BB * HH * LL, 128, 0, stream>>>(S);
    gemm(S, vp, nullptr, nullptr, ctxp, LL, DHD, LL, LL, DHD, DHD,
         (long long)LL * LL, (long long)LL * DHD, (long long)LL * DHD, 0,
         0, 1.0f, 0, BB * HH);
    unpack_ctx_k<<<blkBL, 256, 0, stream>>>(ctxp, ctx);
    // h1 = LN(h + ctx@Wo + bo)
    gemm(ctx, PP(4), PP(5), h, tmp, MTOK, DD, DD, DD, DD, DD,
         0, 0, 0, 0, 0, 1.0f, FLAG_BIAS | FLAG_RES, 1);
    layernorm768<<<MTOK, 256, 0, stream>>>(tmp, PP(6), PP(7), h1);

    // ---- res = gelu(h1 @ m3 + b) ----
    gemm(h1, PP(20), PP(21), nullptr, res, MTOK, DD, DD, DD, DD, DD,
         0, 0, 0, 0, 0, 1.0f, FLAG_BIAS | FLAG_GELU, 1);

    // ---- x = gelu(LN(h1;n1) @ m1 + b) ----
    layernorm768<<<MTOK, 256, 0, stream>>>(h1, PP(8), PP(9), tln);
    gemm(tln, PP(10), PP(11), nullptr, x1, MTOK, DD, DD, DD, DD, DD,
         0, 0, 0, 0, 0, 1.0f, FLAG_BIAS | FLAG_GELU, 1);

    // ---- two GCN layers ----
    auto gcn = [&](const float* xin, const float* W, const float* bsc, float* xout) {
      gemm(xin, W, nullptr, nullptr, xw, MTOK, DD, DD, DD, DD, DD,
           0, 0, 0, 0, 0, 1.0f, 0, 1);
      gcn_self_k<<<blkBL, 256, 0, stream>>>(xw, dinv, xout);
      long long ne = (long long)BB * EE * (DD / 4);
      gcn_edge_k<<<(unsigned)((ne + 255) / 256), 256, 0, stream>>>(xw, ei, dinv, xout);
      bias_gelu_k<<<blkBL, 256, 0, stream>>>(xout, bsc, xout, nBL, DD);
    };
    gcn(x1, PP(12), PP(13), x2);
    gcn(x2, PP(14), PP(15), x1);

    // ---- x = gelu(LN(x;n2) @ m2 + b); g = res + x ----
    layernorm768<<<MTOK, 256, 0, stream>>>(x1, PP(16), PP(17), tln);
    gemm(tln, PP(18), PP(19), nullptr, x2, MTOK, DD, DD, DD, DD, DD,
         0, 0, 0, 0, 0, 1.0f, FLAG_BIAS | FLAG_GELU, 1);
    add_k<<<blkBL, 256, 0, stream>>>(res, x2, g, nBL);

    // ---- FFN + final LN ----
    gemm(g, PP(22), PP(23), nullptr, ffm, MTOK, DFF, DD, DD, DFF, DFF,
         0, 0, 0, 0, 0, 1.0f, FLAG_BIAS | FLAG_GELU, 1);
    gemm(ffm, PP(24), PP(25), g, tmp, MTOK, DD, DFF, DFF, DD, DD,
         0, 0, 0, 0, 0, 1.0f, FLAG_BIAS | FLAG_RES, 1);
    layernorm768<<<MTOK, 256, 0, stream>>>(tmp, PP(26), PP(27), h);
  }

  hipMemcpyAsync(d_out, h, (size_t)nBL * sizeof(float), hipMemcpyDeviceToDevice, stream);
}
